// GQA_137438954255
// MI455X (gfx1250) — matrix operand order repaired and checked by reading, not compiled, not run on hardware
//
#include <hip/hip_runtime.h>
#include <hip/hip_bf16.h>

// ---------------- types ----------------
typedef __attribute__((ext_vector_type(16))) __bf16 v16bf;
typedef __attribute__((ext_vector_type(8)))  float  v8f;
typedef __attribute__((ext_vector_type(4)))  unsigned int u32x4;
typedef __attribute__((ext_vector_type(8)))  int  i32x8;
typedef unsigned int   u32;
typedef unsigned short u16;

union AFrag { v16bf v; u32 u[8]; };

#define TSEQ   2048
#define CDIM   2048
#define HDIM   128
#define NHEAD  16
#define NKV    4

__device__ __forceinline__ u16 f2bf(float x) {
    u32 u = __float_as_uint(x);
    u += 0x7FFFu + ((u >> 16) & 1u);        // round-to-nearest-even
    return (u16)(u >> 16);
}

__device__ __forceinline__ v8f wmma_bf16(const v16bf& a, const v16bf& b, const v8f& c) {
    return __builtin_amdgcn_wmma_f32_16x16x32_bf16(false, a, false, b, (short)0, c, false, false);
}

// Fragment loaders: force 128-bit loads (p must be 16B-aligned).
// gap8: ISA 16-bit A-matrix layout -> uints [0..3] and [8..11]
__device__ __forceinline__ void load_frag_gap8(AFrag& f, const u32* p) {
    uint4 a = *reinterpret_cast<const uint4*>(p);
    uint4 b = *reinterpret_cast<const uint4*>(p + 8);
    f.u[0] = a.x; f.u[1] = a.y; f.u[2] = a.z; f.u[3] = a.w;
    f.u[4] = b.x; f.u[5] = b.y; f.u[6] = b.z; f.u[7] = b.w;
}
// seq: contiguous k order (k = 16h + s) -> uints [0..7]
__device__ __forceinline__ void load_frag_seq(AFrag& f, const u32* p) {
    uint4 a = *reinterpret_cast<const uint4*>(p);
    uint4 b = *reinterpret_cast<const uint4*>(p + 4);
    f.u[0] = a.x; f.u[1] = a.y; f.u[2] = a.z; f.u[3] = a.w;
    f.u[4] = b.x; f.u[5] = b.y; f.u[6] = b.z; f.u[7] = b.w;
}

// ---- CDNA5 async global->LDS (ASYNCcnt) ----
__device__ __forceinline__ void async_copy_b128(void* lds, const void* g) {
    unsigned l = (unsigned)(reinterpret_cast<uintptr_t>(lds));
    asm volatile("global_load_async_to_lds_b128 %0, %1, off"
                 :: "v"(l), "v"(g) : "memory");
}
__device__ __forceinline__ void wait_async0() {
    asm volatile("s_wait_asynccnt 0" ::: "memory");
}
__device__ __forceinline__ void wait_async8() {
    asm volatile("s_wait_asynccnt 8" ::: "memory");
}

// ---- CDNA5 Tensor Data Mover: 2D tile, global -> LDS (TENSORcnt) ----
__device__ __forceinline__ void tdm_load_2d(void* lds, const void* gaddr,
                                            unsigned tile_d0, unsigned tile_d1,
                                            unsigned long long stride0_elems) {
    unsigned long long ga = (unsigned long long)(uintptr_t)gaddr;
    u32x4 g0;
    g0.x = 1u;                                            // count=1, user D#
    g0.y = (unsigned)(uintptr_t)lds;                      // lds_addr (bytes)
    g0.z = (unsigned)(ga & 0xFFFFFFFFu);                  // global_addr[31:0]
    g0.w = (unsigned)((ga >> 32) & 0x01FFFFFFu) | (2u << 30); // addr[56:32]|type=2
    i32x8 g1;
    g1[0] = (int)(1u << 16);                              // wg_mask=0, data_size=2B
    g1[1] = (int)((tile_d0 & 0xFFFFu) << 16);             // tensor_dim0[15:0]
    g1[2] = (int)(((tile_d0 >> 16) & 0xFFFFu) |
                  ((tile_d1 & 0xFFFFu) << 16));           // tensor_dim1[15:0]
    g1[3] = (int)(((tile_d1 >> 16) & 0xFFFFu) |
                  ((tile_d0 & 0xFFFFu) << 16));           // tile_dim0
    g1[4] = (int)(tile_d1 & 0xFFFFu);                     // tile_dim1; tile_dim2=0
    g1[5] = (int)(stride0_elems & 0xFFFFFFFFull);         // dim0_stride[31:0]
    g1[6] = (int)((stride0_elems >> 32) & 0xFFFFull);     // dim0_stride[47:32]
    g1[7] = 0;
    asm volatile("tensor_load_to_lds %0, %1" :: "s"(g0), "s"(g1) : "memory");
}
__device__ __forceinline__ void wait_tensor0() {
    asm volatile("s_wait_tensorcnt 0x0" ::: "memory");
}
__device__ __forceinline__ void wait_tensor2() {
    asm volatile("s_wait_tensorcnt 0x2" ::: "memory");
}

// ---------------- kernel 1: fp32 -> bf16 cast ----------------
__global__ __launch_bounds__(256) void cast_f32_bf16(const float* __restrict__ in,
                                                     u16* __restrict__ out, int n) {
    int i = blockIdx.x * 256 + threadIdx.x;
    if (i < n) out[i] = f2bf(in[i]);
}

// ---------------- kernel 2/5: C[M,N] = A[M,K] * B[N,K]^T, bf16 in, f32 out ----
// 256 thr (8 waves); tile 128x128, BK=64; TDM-staged, double-buffered.
__global__ __launch_bounds__(256) void gemm_bf16_tn(const u16* __restrict__ A,
                                                    const u16* __restrict__ Bw,
                                                    float* __restrict__ C,
                                                    int M, int N, int K) {
    __shared__ __align__(16) u32 sA[2][128 * 32];  // buf x (128 rows x 64 bf16)
    __shared__ __align__(16) u32 sB[2][128 * 32];
    const int tid  = threadIdx.x;
    const int w    = tid >> 5;
    const int lane = tid & 31;
    const int m16  = lane & 15;
    const int hiA  = (lane >> 4) ? 4 : 0;
    const int hiB  = (lane >> 4) ? 8 : 0;
    const int rhi  = (lane >> 4) ? 8 : 0;
    const int wr   = w >> 2;                       // 0..1 : 64-row half
    const int wc   = w & 3;                        // 0..3 : 32-col quarter
    const int tileM = blockIdx.y * 128;
    const int tileN = blockIdx.x * 128;

    v8f acc[4][2] = {};

    auto stage = [&](int bufi, int k0) {           // one wave: 2 TDM descriptors
        tdm_load_2d(&sA[bufi][0], A  + (size_t)tileM * K + k0, 64, 128,
                    (unsigned long long)K);
        tdm_load_2d(&sB[bufi][0], Bw + (size_t)tileN * K + k0, 64, 128,
                    (unsigned long long)K);
    };

    if (w == 0) stage(0, 0);
    int buf = 0;
    for (int k0 = 0; k0 < K; k0 += 64) {
        if (w == 0) {                  // DMA next tile while WMMA runs on current
            if (k0 + 64 < K) { stage(buf ^ 1, k0 + 64); wait_tensor2(); }
            else             { wait_tensor0(); }
        }
        __syncthreads();

#pragma unroll
        for (int kc = 0; kc < 2; ++kc) {           // two K=32 chunks
            AFrag afr[4];
#pragma unroll
            for (int i = 0; i < 4; ++i)
                load_frag_gap8(afr[i],
                    &sA[buf][(wr * 64 + i * 16 + m16) * 32 + kc * 16 + hiA]);
#pragma unroll
            for (int bc = 0; bc < 2; ++bc) {
                AFrag bfr;
                load_frag_gap8(bfr,
                    &sB[buf][(wc * 32 + bc * 16 + m16) * 32 + kc * 16 + hiA]);
#pragma unroll
                for (int i = 0; i < 4; ++i)
                    acc[i][bc] = wmma_bf16(afr[i].v, bfr.v, acc[i][bc]);
            }
        }
        __syncthreads();
        buf ^= 1;
    }

    const int colbase = tileN + wc * 32 + m16;
#pragma unroll
    for (int i = 0; i < 4; ++i) {
        const int rbase = tileM + wr * 64 + i * 16 + rhi;
#pragma unroll
        for (int bc = 0; bc < 2; ++bc)
#pragma unroll
            for (int r = 0; r < 8; ++r)
                C[(size_t)(rbase + r) * N + colbase + bc * 16] = acc[i][bc][r];
    }
}

// ---------------- kernel 3: QK rms-norm + RoPE + layout shuffle ----------------
__device__ __forceinline__ float block_sum128(float v, float* red, int tid) {
#pragma unroll
    for (int m = 16; m >= 1; m >>= 1) v += __shfl_xor(v, m, 32);
    if ((tid & 31) == 0) red[tid >> 5] = v;
    __syncthreads();
    float tot = red[0] + red[1] + red[2] + red[3];
    __syncthreads();
    return tot;
}

__global__ __launch_bounds__(128) void rope_norm_kernel(
        const float* __restrict__ Qf, const float* __restrict__ Kf,
        const float* __restrict__ Vf,
        const float* __restrict__ qn_w, const float* __restrict__ kn_w,
        const float* __restrict__ cosT, const float* __restrict__ sinT,
        u16* __restrict__ Qb, u16* __restrict__ Kb, u16* __restrict__ Vb) {
    __shared__ float red[4];
    const int d = threadIdx.x;               // 0..127
    const int t = blockIdx.x;
    const int b = blockIdx.y;
    const int tid = threadIdx.x;
    const float c = cosT[t * HDIM + d];
    const float s = sinT[t * HDIM + d];
    const int dpair = (d < 64) ? d + 64 : d - 64;
    const float sgn = (d < 64) ? -1.f : 1.f;

    for (int h = 0; h < NHEAD; ++h) {
        const float* q = Qf + ((size_t)(b * TSEQ + t)) * CDIM + h * HDIM;
        float v = q[d];
        float tot = block_sum128(v * v, red, tid);
        float inv = rsqrtf(tot / (float)HDIM + 1e-6f);
        float vn = v * inv * qn_w[d];
        float vo = q[dpair] * inv * qn_w[dpair];
        Qb[((size_t)(b * NHEAD + h) * TSEQ + t) * HDIM + d] = f2bf(vn * c + sgn * vo * s);
    }
    for (int h = 0; h < NKV; ++h) {
        const float* k = Kf + ((size_t)(b * TSEQ + t)) * (NKV * HDIM) + h * HDIM;
        float v = k[d];
        float tot = block_sum128(v * v, red, tid);
        float inv = rsqrtf(tot / (float)HDIM + 1e-6f);
        float vn = v * inv * kn_w[d];
        float vo = k[dpair] * inv * kn_w[dpair];
        Kb[((size_t)(b * NKV + h) * TSEQ + t) * HDIM + d] = f2bf(vn * c + sgn * vo * s);
    }
    for (int h = 0; h < NKV; ++h) {   // V: cast + transpose to [B,KV,D,T]
        float v = Vf[((size_t)(b * TSEQ + t)) * (NKV * HDIM) + h * HDIM + d];
        Vb[((size_t)(b * NKV + h) * HDIM + d) * TSEQ + t] = f2bf(v);
    }
}

// ---------------- kernel 4: flash attention ----------------
// 128 thr (4 waves); grid (T/64, NHEAD, B); async K/V staging, double-buffered.
__global__ __launch_bounds__(128) void flash_attn_kernel(
        const u16* __restrict__ Qb, const u16* __restrict__ Kb,
        const u16* __restrict__ Vb, u16* __restrict__ Y) {
    __shared__ __align__(16) u32 sK[2][32 * 64];   // 32 keys x 128 bf16
    __shared__ __align__(16) u32 sV[2][128 * 16];  // 128 dims x 32 keys
    __shared__ __align__(16) u32 sP[4 * 16 * 16];  // per-wave 16 x 32 bf16 P tile

    const int b   = blockIdx.z;
    const int h   = blockIdx.y;
    const int qb  = blockIdx.x * 64;
    const int kvh = h >> 2;                        // G = 4
    const int tid = threadIdx.x;
    const int w   = tid >> 5;
    const int lane = tid & 31;
    const int m16 = lane & 15;
    const int hiA = (lane >> 4) ? 4 : 0;
    const int hiB = (lane >> 4) ? 8 : 0;
    const int rhi = (lane >> 4) ? 8 : 0;

    const u16* Kbase = Kb + (size_t)(b * NKV + kvh) * TSEQ * HDIM;
    const u16* Vbase = Vb + ((size_t)(b * NKV + kvh) * HDIM + tid) * TSEQ;

    auto stage = [&](int bufi, int kb) {           // 8 async b128 per thread
        const u16* gK = Kbase + (size_t)kb * HDIM + tid * 32;
        u32* lK = &sK[bufi][tid * 16];
        const u16* gV = Vbase + kb;
        u32* lV = &sV[bufi][tid * 16];
#pragma unroll
        for (int i = 0; i < 4; ++i) {
            async_copy_b128(lK + i * 4, gK + i * 8);
            async_copy_b128(lV + i * 4, gV + i * 8);
        }
    };

    const u32* Qu = reinterpret_cast<const u32*>(
        Qb + ((size_t)(b * NHEAD + h) * TSEQ + qb + w * 16) * HDIM) + m16 * 64;
    AFrag qf[4];
#pragma unroll
    for (int c2 = 0; c2 < 4; ++c2)
        load_frag_gap8(qf[c2], Qu + c2 * 16 + hiA);

    v8f acc[8] = {};
    float mrow[8], lrow[8];
#pragma unroll
    for (int r = 0; r < 8; ++r) { mrow[r] = -1e30f; lrow[r] = 0.f; }

    const float scale = 0.08838834764831845f;      // 1/sqrt(128)
    const int kend = qb + 64;

    stage(0, 0);
    int buf = 0;
    for (int kb = 0; kb < kend; kb += 32) {
        if (kb + 32 < kend) { stage(buf ^ 1, kb + 32); wait_async8(); }
        else                { wait_async0(); }
        __syncthreads();

        // S = Q * K^T : two key groups of 16 -> s0, s1
        v8f s0 = {}, s1 = {};
#pragma unroll
        for (int c2 = 0; c2 < 4; ++c2) {
            AFrag b0, b1;
            load_frag_gap8(b0, &sK[buf][m16 * 64 + c2 * 16 + hiA]);
            load_frag_gap8(b1, &sK[buf][(m16 + 16) * 64 + c2 * 16 + hiA]);
            s0 = wmma_bf16(qf[c2].v, b0.v, s0);
            s1 = wmma_bf16(qf[c2].v, b1.v, s1);
        }

        // online softmax
#pragma unroll
        for (int r = 0; r < 8; ++r) {
            const int qidx = qb + w * 16 + r + rhi;
            float a0 = s0[r] * scale, a1 = s1[r] * scale;
            if (kb + m16      > qidx) a0 = -1e30f;
            if (kb + 16 + m16 > qidx) a1 = -1e30f;
            float mx = fmaxf(a0, a1);
#pragma unroll
            for (int msk = 8; msk >= 1; msk >>= 1) mx = fmaxf(mx, __shfl_xor(mx, msk, 16));
            float mnew  = fmaxf(mrow[r], mx);
            float alpha = __expf(mrow[r] - mnew);
            mrow[r] = mnew;
            float p0 = __expf(a0 - mnew), p1 = __expf(a1 - mnew);
            float ps = p0 + p1;
#pragma unroll
            for (int msk = 8; msk >= 1; msk >>= 1) ps += __shfl_xor(ps, msk, 16);
            lrow[r] = lrow[r] * alpha + ps;
#pragma unroll
            for (int a = 0; a < 8; ++a) acc[a][r] *= alpha;
            u16* pr = reinterpret_cast<u16*>(&sP[(w * 16 + r + rhi) * 16]);
            pr[m16]      = f2bf(p0);
            pr[m16 + 16] = f2bf(p1);
        }
        asm volatile("s_wait_dscnt 0" ::: "memory");   // wave-private LDS RAW

        AFrag pf;
        load_frag_gap8(pf, &sP[(w * 16 + m16) * 16 + hiA]);
#pragma unroll
        for (int a = 0; a < 8; ++a) {
            AFrag vfr;
            load_frag_gap8(vfr, &sV[buf][(a * 16 + m16) * 16 + hiA]);
            acc[a] = wmma_bf16(pf.v, vfr.v, acc[a]);
        }
        __syncthreads();
        buf ^= 1;
    }

#pragma unroll
    for (int a = 0; a < 8; ++a) {
#pragma unroll
        for (int r = 0; r < 8; ++r) {
            const int row = qb + w * 16 + r + rhi;
            float o = acc[a][r] / lrow[r];
            Y[((size_t)(b * TSEQ + row)) * CDIM + h * HDIM + a * 16 + m16] = f2bf(o);
        }
    }
}

// ---------------- launcher ----------------
extern "C" void kernel_launch(void* const* d_in, const int* in_sizes, int n_in,
                              void* d_out, int out_size, void* d_ws, size_t ws_size,
                              hipStream_t stream) {
    const float* x    = (const float*)d_in[0];
    const float* Wq   = (const float*)d_in[1];
    const float* Wk   = (const float*)d_in[2];
    const float* Wv   = (const float*)d_in[3];
    const float* Wo   = (const float*)d_in[4];
    const float* qnw  = (const float*)d_in[5];
    const float* knw  = (const float*)d_in[6];
    const float* rc   = (const float*)d_in[7];
    const float* rs   = (const float*)d_in[8];

    const int B = 2, M = B * TSEQ;               // 4096 rows
    const size_t nX  = (size_t)M * CDIM;         // 8M
    const size_t nWq = (size_t)CDIM * CDIM;      // 4M
    const size_t nWk = (size_t)(NKV * HDIM) * CDIM; // 1M

    char* ws = (char*)d_ws;
    size_t off = 0;
    auto alloc = [&](size_t bytes) { size_t o = off; off += (bytes + 255) & ~(size_t)255; return o; };
    u16*   xb  = (u16*)  (ws + alloc(nX  * 2));
    u16*   wqb = (u16*)  (ws + alloc(nWq * 2));
    u16*   wkb = (u16*)  (ws + alloc(nWk * 2));
    u16*   wvb = (u16*)  (ws + alloc(nWk * 2));
    u16*   wob = (u16*)  (ws + alloc(nWq * 2));
    float* qf  = (float*)(ws + alloc(nX * 4));              // reused as Y region later
    float* kf  = (float*)(ws + alloc((size_t)M * NKV * HDIM * 4));
    float* vf  = (float*)(ws + alloc((size_t)M * NKV * HDIM * 4));
    u16*   qb2 = (u16*)  (ws + alloc(nX * 2));
    u16*   kb2 = (u16*)  (ws + alloc((size_t)M * NKV * HDIM * 2));
    u16*   vb2 = (u16*)  (ws + alloc((size_t)M * NKV * HDIM * 2));
    u16*   yb  = (u16*)qf;   // alias: qf fp32 dead after rope_norm; Y bf16 fits

    cast_f32_bf16<<<(int)((nX  + 255) / 256), 256, 0, stream>>>(x,  xb,  (int)nX);
    cast_f32_bf16<<<(int)((nWq + 255) / 256), 256, 0, stream>>>(Wq, wqb, (int)nWq);
    cast_f32_bf16<<<(int)((nWk + 255) / 256), 256, 0, stream>>>(Wk, wkb, (int)nWk);
    cast_f32_bf16<<<(int)((nWk + 255) / 256), 256, 0, stream>>>(Wv, wvb, (int)nWk);
    cast_f32_bf16<<<(int)((nWq + 255) / 256), 256, 0, stream>>>(Wo, wob, (int)nWq);

    gemm_bf16_tn<<<dim3(CDIM / 128, M / 128), 256, 0, stream>>>(xb, wqb, qf, M, CDIM, CDIM);
    gemm_bf16_tn<<<dim3((NKV * HDIM) / 128, M / 128), 256, 0, stream>>>(xb, wkb, kf, M, NKV * HDIM, CDIM);
    gemm_bf16_tn<<<dim3((NKV * HDIM) / 128, M / 128), 256, 0, stream>>>(xb, wvb, vf, M, NKV * HDIM, CDIM);

    rope_norm_kernel<<<dim3(TSEQ, B), 128, 0, stream>>>(qf, kf, vf, qnw, knw, rc, rs,
                                                        qb2, kb2, vb2);

    flash_attn_kernel<<<dim3(TSEQ / 64, NHEAD, B), 128, 0, stream>>>(qb2, kb2, vb2, yb);

    gemm_bf16_tn<<<dim3(CDIM / 128, M / 128), 256, 0, stream>>>(yb, wob, (float*)d_out,
                                                                M, CDIM, CDIM);
}
